// Gatconv_22213570855008
// MI455X (gfx1250) — compile-verified, matrix operand order checked by reading, not run on hardware
//
#include <hip/hip_runtime.h>
#include <hip/hip_bf16.h>
#include <stdint.h>

// ---------------------------------------------------------------------------
// Types for CDNA5 WMMA
// ---------------------------------------------------------------------------
typedef __attribute__((ext_vector_type(16))) __bf16 v16bf;
typedef __attribute__((ext_vector_type(8)))  float  v8f;

union BfVec {
  v16bf v;
  __bf16 e[16];
  unsigned int u[8];
};

#define NN   50000
#define INF_ 256
#define HIDF 128
#define EE   640000
#define L1F  200     // 50000 = 200 * 250
#define L2F  250
#define TOPK 10
#define PI_D 3.14159265358979323846

// ---------------------------------------------------------------------------
// Utility kernels
// ---------------------------------------------------------------------------
__global__ void k_fill_f32(float* __restrict__ p, int n, float v) {
  int i = blockIdx.x * blockDim.x + threadIdx.x;
  if (i < n) p[i] = v;
}

__global__ void k_cvt_bf16(const float* __restrict__ s, __bf16* __restrict__ d, int n) {
  int i = blockIdx.x * blockDim.x + threadIdx.x;
  if (i < n) d[i] = (__bf16)s[i];
}

// out[i] = bf16( x[i] * rsqrt(max(deg[row],1)) )
__global__ void k_scale_to_bf16(const float* __restrict__ x, const float* __restrict__ deg,
                                __bf16* __restrict__ out, int n, int cols) {
  int i = blockIdx.x * blockDim.x + threadIdx.x;
  if (i >= n) return;
  int row = i / cols;
  float s = rsqrtf(fmaxf(deg[row], 1.0f));
  out[i] = (__bf16)(x[i] * s);
}

__global__ void k_edge_degree(const int* __restrict__ src, const int* __restrict__ dst,
                              float* __restrict__ degOut, float* __restrict__ degIn, int E) {
  int i = blockIdx.x * blockDim.x + threadIdx.x;
  if (i >= E) return;
  atomicAdd(&degOut[src[i]], 1.0f);
  atomicAdd(&degIn[dst[i]], 1.0f);
}

// agg[dst[e]][c] += h[src[e]][c]
__global__ __launch_bounds__(128)
void k_scatter_edges(const float* __restrict__ h, const int* __restrict__ src,
                     const int* __restrict__ dst, float* __restrict__ agg) {
  int e = blockIdx.x;
  int c = threadIdx.x;
  int s = src[e], d = dst[e];
  atomicAdd(&agg[(size_t)d * HIDF + c], h[(size_t)s * HIDF + c]);
}

// h = act(agg * rsqrt(max(degIn,1)) + bias) -> bf16
__global__ void k_postagg_bf16(const float* __restrict__ agg, const float* __restrict__ deg,
                               const float* __restrict__ bias, __bf16* __restrict__ out,
                               int n, int relu) {
  int i = blockIdx.x * blockDim.x + threadIdx.x;
  if (i >= n) return;
  int row = i / HIDF, col = i % HIDF;
  float s = rsqrtf(fmaxf(deg[row], 1.0f));
  float v = agg[i] * s + bias[col];
  if (relu) v = fmaxf(v, 0.0f);
  out[i] = (__bf16)v;
}

__global__ void k_postagg_f32(const float* __restrict__ agg, const float* __restrict__ deg,
                              const float* __restrict__ bias, float* __restrict__ out, int n) {
  int i = blockIdx.x * blockDim.x + threadIdx.x;
  if (i >= n) return;
  int row = i / HIDF, col = i % HIDF;
  float s = rsqrtf(fmaxf(deg[row], 1.0f));
  out[i] = agg[i] * s + bias[col];
}

// ---------------------------------------------------------------------------
// Weight pack: convert f32 W[K x 128] -> bf16 WMMA B-fragment order.
// P[((kt*8 + nt)*32 + lane)*8 + p] = pack( W[k0+kb][n], W[k0+kb+1][n] )
// so each lane of the GEMM wave reads its whole fragment as 32 contiguous B.
// ---------------------------------------------------------------------------
__global__ void k_pack_w(const float* __restrict__ W, unsigned int* __restrict__ P, int K) {
  int total = (K >> 5) * 2048;   // (K/32) * 8 tiles * 32 lanes * 8 dwords
  int i = blockIdx.x * blockDim.x + threadIdx.x;
  if (i >= total) return;
  int pI   = i & 7;
  int lane = (i >> 3) & 31;
  int nt   = (i >> 8) & 7;
  int kt   = i >> 11;
  int r = lane & 15, hi = lane >> 4;
  int kb = (pI < 4 ? 2 * pI : 2 * (pI - 4) + 16) + (hi << 3);
  int k0 = kt * 32;
  int n = nt * 16 + r;
  __bf16 lo = (__bf16)W[(size_t)(k0 + kb) * HIDF + n];
  __bf16 hiE = (__bf16)W[(size_t)(k0 + kb + 1) * HIDF + n];
  unsigned short ul, uh;
  __builtin_memcpy(&ul, &lo, 2);
  __builtin_memcpy(&uh, &hiE, 2);
  P[i] = (unsigned int)ul | ((unsigned int)uh << 16);
}

// ---------------------------------------------------------------------------
// WMMA bf16 GEMM:  C[M x 128] = A[M x K] @ B[K x 128] (+ bias)
// One wave32 per 16-row block; 8 column tiles; B pre-packed in fragment order.
// Per K-tile: A fragment (2x b128) + all 8 B fragments (16x b128) are issued
// first; a sched_barrier pins them before the WMMA burst so the 8 fragments
// stay in distinct VGPRs and the WMMAs drain with partial s_wait_loadcnt
// (load/WMMA overlap within the wave).
// ---------------------------------------------------------------------------
__global__ __launch_bounds__(32)
void k_gemm_bf16(const __bf16* __restrict__ A, const unsigned int* __restrict__ Bp,
                 float* __restrict__ C, const float* __restrict__ bias, int K) {
  const int m0 = blockIdx.x * 16;
  const int lane = threadIdx.x;
  const int r = lane & 15;      // row (A) / col (C)
  const int hi = lane >> 4;     // K-half selector

  v8f acc[8];
#pragma unroll
  for (int nt = 0; nt < 8; ++nt) {
    v8f z = {0.f, 0.f, 0.f, 0.f, 0.f, 0.f, 0.f, 0.f};
    acc[nt] = z;
  }

  const v16bf* Bf = (const v16bf*)Bp;
  const int kTiles = K >> 5;
  for (int kt = 0; kt < kTiles; ++kt) {
    const __bf16* arow = A + (size_t)(m0 + r) * K + (kt << 5);
    if (kt + 1 < kTiles) __builtin_prefetch(arow + 32, 0, 3);  // near-scope
    BfVec a;
#pragma unroll
    for (int p = 0; p < 8; ++p) {
      int kb = (p < 4 ? 2 * p : 2 * (p - 4) + 16) + (hi << 3);
      a.u[p] = *(const unsigned int*)(arow + kb);  // 4B-aligned bf16 pair
    }
    v16bf bfrag[8];
#pragma unroll
    for (int nt = 0; nt < 8; ++nt)
      bfrag[nt] = Bf[((size_t)(kt * 8 + nt)) * 32 + lane];
    // Keep all fragment loads issued (in flight) before any WMMA consumes:
    __builtin_amdgcn_sched_barrier(0);
#pragma unroll
    for (int nt = 0; nt < 8; ++nt)
      acc[nt] = __builtin_amdgcn_wmma_f32_16x16x32_bf16(
          false, a.v, false, bfrag[nt], (short)0, acc[nt], false, false);
  }

#pragma unroll
  for (int nt = 0; nt < 8; ++nt) {
    int col = nt * 16 + r;
    float bv = bias ? bias[col] : 0.0f;
#pragma unroll
    for (int j = 0; j < 8; ++j) {
      int row = m0 + j + (hi << 3);
      C[(size_t)row * HIDF + col] = acc[nt][j] + bv;
    }
  }
}

// ---------------------------------------------------------------------------
// Four-step DFT machinery (L = 50000 = 200*250), DFT matrices are symmetric.
// Tables store cos/sin of -2*pi*j*k/denom.
// ---------------------------------------------------------------------------
__global__ void k_build_table(float* __restrict__ R, float* __restrict__ I,
                              int rows, int cols, int denom) {
  int i = blockIdx.x * blockDim.x + threadIdx.x;
  if (i >= rows * cols) return;
  int j = i / cols, k = i % cols;
  double ang = -2.0 * PI_D * (double)((long long)j * k) / (double)denom;
  R[i] = (float)cos(ang);
  I[i] = (float)sin(ang);
}

// Stage 1 (real input):  Y[a][c] = sum_b x[a+200b, ch] * W250^{cb}
// grid (200, 8) = (a, cc); 250 active threads = c
__global__ __launch_bounds__(256)
void k_dft1_real(const float* __restrict__ x, int ch0,
                 const float* __restrict__ tR, const float* __restrict__ tI,
                 float* __restrict__ Yr, float* __restrict__ Yi) {
  __shared__ float xs[L2F];
  int a = blockIdx.x, cc = blockIdx.y, ch = ch0 + cc;
  int t = threadIdx.x;
  if (t < L2F) xs[t] = x[(size_t)(a + L1F * t) * HIDF + ch];
  __syncthreads();
  if (t >= L2F) return;
  float yr = 0.f, yi = 0.f;
  for (int b = 0; b < L2F; ++b) {
    float xv = xs[b];
    yr += xv * tR[b * L2F + t];   // symmetric DFT matrix -> coalesced
    yi += xv * tI[b * L2F + t];
  }
  size_t o = ((size_t)cc * L1F + a) * L2F + t;
  Yr[o] = yr; Yi[o] = yi;
}

// Stage 2: F[c + 250 d] = sum_a W200^{da} * ( W_L^{ca} * Y[a][c] )
// grid (200, 8) = (d, cc)
__global__ __launch_bounds__(256)
void k_dft2(const float* __restrict__ Yr, const float* __restrict__ Yi,
            const float* __restrict__ twR, const float* __restrict__ twI,
            const float* __restrict__ d2R, const float* __restrict__ d2I,
            float* __restrict__ Fr, float* __restrict__ Fi) {
  int d = blockIdx.x, cc = blockIdx.y;
  int c = threadIdx.x;
  if (c >= L2F) return;
  float fr = 0.f, fi = 0.f;
  for (int a = 0; a < L1F; ++a) {
    size_t yo = ((size_t)cc * L1F + a) * L2F + c;
    float yr = Yr[yo], yi = Yi[yo];
    float tr = twR[a * L2F + c], ti = twI[a * L2F + c];
    float zr = yr * tr - yi * ti;
    float zi = yr * ti + yi * tr;
    float wr = d2R[d * L1F + a], wi = d2I[d * L1F + a];  // scalar-uniform
    fr += wr * zr - wi * zi;
    fi += wr * zi + wi * zr;
  }
  size_t f = (size_t)cc * NN + c + L2F * d;
  Fr[f] = fr; Fi[f] = fi;
}

// S[f] += sum_cc Fq[cc][f] * conj(Fk[cc][f])
__global__ void k_xspec(const float* __restrict__ Qr, const float* __restrict__ Qi,
                        const float* __restrict__ Kr, const float* __restrict__ Ki,
                        float* __restrict__ Sr, float* __restrict__ Si) {
  int f = blockIdx.x * blockDim.x + threadIdx.x;
  if (f >= NN) return;
  float sr = 0.f, si = 0.f;
#pragma unroll
  for (int cc = 0; cc < 8; ++cc) {
    size_t o = (size_t)cc * NN + f;
    float qr = Qr[o], qi = Qi[o], kr = Kr[o], ki = Ki[o];
    sr += qr * kr + qi * ki;
    si += qi * kr - qr * ki;
  }
  Sr[f] += sr; Si[f] += si;
}

// Inverse stage 1 (complex input, conjugate tables)
__global__ __launch_bounds__(256)
void k_idft1(const float* __restrict__ Sr, const float* __restrict__ Si,
             const float* __restrict__ tR, const float* __restrict__ tI,
             float* __restrict__ Yr, float* __restrict__ Yi) {
  __shared__ float shr[L2F], shi[L2F];
  int a = blockIdx.x, t = threadIdx.x;
  if (t < L2F) { shr[t] = Sr[a + L1F * t]; shi[t] = Si[a + L1F * t]; }
  __syncthreads();
  if (t >= L2F) return;
  float yr = 0.f, yi = 0.f;
  for (int b = 0; b < L2F; ++b) {
    float tr = tR[b * L2F + t];
    float ti = -tI[b * L2F + t];     // conj
    float xr = shr[b], xi = shi[b];
    yr += xr * tr - xi * ti;
    yi += xr * ti + xi * tr;
  }
  Yr[a * L2F + t] = yr; Yi[a * L2F + t] = yi;
}

// Inverse stage 2 -> corr (real part, scaled by 1/(128 * L))
__global__ __launch_bounds__(256)
void k_idft2(const float* __restrict__ Yr, const float* __restrict__ Yi,
             const float* __restrict__ twR, const float* __restrict__ twI,
             const float* __restrict__ d2R, const float* __restrict__ d2I,
             float* __restrict__ corr) {
  int d = blockIdx.x, c = threadIdx.x;
  if (c >= L2F) return;
  float fr = 0.f;
  for (int a = 0; a < L1F; ++a) {
    float yr = Yr[a * L2F + c], yi = Yi[a * L2F + c];
    float tr = twR[a * L2F + c], ti = -twI[a * L2F + c];
    float zr = yr * tr - yi * ti;
    float zi = yr * ti + yi * tr;
    float wr = d2R[d * L1F + a], wi = -d2I[d * L1F + a];
    fr += wr * zr - wi * zi;
  }
  corr[c + L2F * d] = fr * (1.0f / (128.0f * (float)NN));
}

// Single-block top-10 + softmax over corr[0..NN)
__global__ __launch_bounds__(1024)
void k_topk_softmax(float* __restrict__ corr, int* __restrict__ idxOut,
                    float* __restrict__ wOut) {
  __shared__ float sv[1024];
  __shared__ int   sx[1024];
  __shared__ float vals[TOPK];
  int t = threadIdx.x;
  for (int kk = 0; kk < TOPK; ++kk) {
    float best = -3.0e38f; int bi = 0;
    for (int i = t; i < NN; i += 1024) {
      float v = corr[i];
      if (v > best) { best = v; bi = i; }
    }
    sv[t] = best; sx[t] = bi;
    __syncthreads();
    for (int off = 512; off > 0; off >>= 1) {
      if (t < off) {
        if (sv[t + off] > sv[t] || (sv[t + off] == sv[t] && sx[t + off] < sx[t])) {
          sv[t] = sv[t + off]; sx[t] = sx[t + off];
        }
      }
      __syncthreads();
    }
    if (t == 0) {
      idxOut[kk] = sx[0];
      vals[kk] = sv[0];
      corr[sx[0]] = -3.0e38f;
    }
    __syncthreads();
  }
  if (t == 0) {
    float mx = vals[0];
    for (int k = 1; k < TOPK; ++k) mx = fmaxf(mx, vals[k]);
    float e[TOPK], s = 0.f;
    for (int k = 0; k < TOPK; ++k) { e[k] = expf(vals[k] - mx); s += e[k]; }
    for (int k = 0; k < TOPK; ++k) wOut[k] = e[k] / s;
  }
}

// out[t][c] = sum_i w[i] * v[(t + idx[i]) % N][c]
__global__ __launch_bounds__(128)
void k_rollagg(const float* __restrict__ v, const int* __restrict__ idx,
               const float* __restrict__ w, float* __restrict__ out) {
  __shared__ int   sidx[TOPK];
  __shared__ float sw[TOPK];
  int t = blockIdx.x, c = threadIdx.x;
  if (c < TOPK) { sidx[c] = idx[c]; sw[c] = w[c]; }
  __syncthreads();
  float acc = 0.f;
#pragma unroll
  for (int i = 0; i < TOPK; ++i) {
    int tt = t + sidx[i];
    if (tt >= NN) tt -= NN;
    acc += sw[i] * v[(size_t)tt * HIDF + c];
  }
  out[(size_t)t * HIDF + c] = acc;
}

// ---------------------------------------------------------------------------
// Host launch
// ---------------------------------------------------------------------------
extern "C" void kernel_launch(void* const* d_in, const int* in_sizes, int n_in,
                              void* d_out, int out_size, void* d_ws, size_t ws_size,
                              hipStream_t stream) {
  const float* node_feats = (const float*)d_in[0];
  const int*   src = (const int*)d_in[1];
  const int*   dst = (const int*)d_in[2];
  const float* W1 = (const float*)d_in[3];
  const float* b1 = (const float*)d_in[4];
  const float* Wq = (const float*)d_in[5];
  const float* bq = (const float*)d_in[6];
  const float* Wk = (const float*)d_in[7];
  const float* bk = (const float*)d_in[8];
  const float* Wv = (const float*)d_in[9];
  const float* bv = (const float*)d_in[10];
  const float* Wo = (const float*)d_in[11];
  const float* bo = (const float*)d_in[12];
  const float* W2 = (const float*)d_in[13];
  const float* b2 = (const float*)d_in[14];
  float* out = (float*)d_out;

  // ---- workspace carve-up ----
  uint8_t* p = (uint8_t*)d_ws;
  auto alloc = [&](size_t bytes) -> void* {
    void* r = (void*)p;
    p += (bytes + 255) & ~(size_t)255;
    return r;
  };
  float*  degOut = (float*)alloc((size_t)NN * 4);
  float*  degIn  = (float*)alloc((size_t)NN * 4);
  __bf16* xsb    = (__bf16*)alloc((size_t)NN * INF_ * 2);   // scaled input (bf16)
  unsigned int* pw1 = (unsigned int*)alloc((size_t)(INF_ / 32) * 2048 * 4);
  unsigned int* pwq = (unsigned int*)alloc((size_t)(HIDF / 32) * 2048 * 4);
  unsigned int* pwk = (unsigned int*)alloc((size_t)(HIDF / 32) * 2048 * 4);
  unsigned int* pwv = (unsigned int*)alloc((size_t)(HIDF / 32) * 2048 * 4);
  unsigned int* pwo = (unsigned int*)alloc((size_t)(HIDF / 32) * 2048 * 4);
  unsigned int* pw2 = (unsigned int*)alloc((size_t)(HIDF / 32) * 2048 * 4);
  float*  bufA   = (float*)alloc((size_t)NN * HIDF * 4);    // hW -> outAC -> t2
  float*  bufB   = (float*)alloc((size_t)NN * HIDF * 4);    // agg -> h2
  __bf16* hb     = (__bf16*)alloc((size_t)NN * HIDF * 2);   // h (bf16) -> outAC (bf16)
  float*  q      = (float*)alloc((size_t)NN * HIDF * 4);
  float*  kk     = (float*)alloc((size_t)NN * HIDF * 4);    // k -> agg2
  float*  v      = (float*)alloc((size_t)NN * HIDF * 4);
  float*  d250R  = (float*)alloc((size_t)L2F * L2F * 4);
  float*  d250I  = (float*)alloc((size_t)L2F * L2F * 4);
  float*  d200R  = (float*)alloc((size_t)L1F * L1F * 4);
  float*  d200I  = (float*)alloc((size_t)L1F * L1F * 4);
  float*  twR    = (float*)alloc((size_t)L1F * L2F * 4);
  float*  twI    = (float*)alloc((size_t)L1F * L2F * 4);
  float*  Yr     = (float*)alloc((size_t)8 * NN * 4);
  float*  Yi     = (float*)alloc((size_t)8 * NN * 4);
  float*  FqR    = (float*)alloc((size_t)8 * NN * 4);
  float*  FqI    = (float*)alloc((size_t)8 * NN * 4);
  float*  FkR    = (float*)alloc((size_t)8 * NN * 4);
  float*  FkI    = (float*)alloc((size_t)8 * NN * 4);
  float*  Sr     = (float*)alloc((size_t)NN * 4);
  float*  Si     = (float*)alloc((size_t)NN * 4);
  float*  Y2r    = (float*)alloc((size_t)NN * 4);
  float*  Y2i    = (float*)alloc((size_t)NN * 4);
  float*  corr   = (float*)alloc((size_t)NN * 4);
  int*    tIdx   = (int*)alloc(64);
  float*  tW     = (float*)alloc(64);

  const int TB = 256;
  auto gb = [&](int n) { return (n + TB - 1) / TB; };
  const int rowBlocks = NN / 16;  // 3125
  const int packN256 = (INF_ / 32) * 2048;   // dwords in packed 256x128 weight
  const int packN128 = (HIDF / 32) * 2048;

  // ---- degrees ----
  k_fill_f32<<<gb(NN), TB, 0, stream>>>(degOut, NN, 0.f);
  k_fill_f32<<<gb(NN), TB, 0, stream>>>(degIn, NN, 0.f);
  k_edge_degree<<<gb(EE), TB, 0, stream>>>(src, dst, degOut, degIn, EE);

  // ---- input conversion + weight packing (f32 -> bf16 fragment order) ----
  k_scale_to_bf16<<<gb(NN * INF_), TB, 0, stream>>>(node_feats, degOut, xsb, NN * INF_, INF_);
  k_pack_w<<<gb(packN256), TB, 0, stream>>>(W1, pw1, INF_);
  k_pack_w<<<gb(packN128), TB, 0, stream>>>(Wq, pwq, HIDF);
  k_pack_w<<<gb(packN128), TB, 0, stream>>>(Wk, pwk, HIDF);
  k_pack_w<<<gb(packN128), TB, 0, stream>>>(Wv, pwv, HIDF);
  k_pack_w<<<gb(packN128), TB, 0, stream>>>(Wo, pwo, HIDF);
  k_pack_w<<<gb(packN128), TB, 0, stream>>>(W2, pw2, HIDF);

  // ---- GraphConv1 ----
  k_gemm_bf16<<<rowBlocks, 32, 0, stream>>>(xsb, pw1, bufA, nullptr, INF_);
  k_fill_f32<<<gb(NN * HIDF), TB, 0, stream>>>(bufB, NN * HIDF, 0.f);
  k_scatter_edges<<<EE, 128, 0, stream>>>(bufA, src, dst, bufB);
  k_postagg_bf16<<<gb(NN * HIDF), TB, 0, stream>>>(bufB, degIn, b1, hb, NN * HIDF, 1);

  // ---- Q/K/V projections ----
  k_gemm_bf16<<<rowBlocks, 32, 0, stream>>>(hb, pwq, q, bq, HIDF);
  k_gemm_bf16<<<rowBlocks, 32, 0, stream>>>(hb, pwk, kk, bk, HIDF);
  k_gemm_bf16<<<rowBlocks, 32, 0, stream>>>(hb, pwv, v, bv, HIDF);

  // ---- DFT tables ----
  k_build_table<<<gb(L2F * L2F), TB, 0, stream>>>(d250R, d250I, L2F, L2F, L2F);
  k_build_table<<<gb(L1F * L1F), TB, 0, stream>>>(d200R, d200I, L1F, L1F, L1F);
  k_build_table<<<gb(L1F * L2F), TB, 0, stream>>>(twR, twI, L1F, L2F, NN);

  // ---- cross-spectrum, chunked over channels (16 chunks of 8) ----
  k_fill_f32<<<gb(NN), TB, 0, stream>>>(Sr, NN, 0.f);
  k_fill_f32<<<gb(NN), TB, 0, stream>>>(Si, NN, 0.f);
  dim3 gDft(L1F, 8);
  for (int chunk = 0; chunk < 16; ++chunk) {
    int ch0 = chunk * 8;
    k_dft1_real<<<gDft, 256, 0, stream>>>(q, ch0, d250R, d250I, Yr, Yi);
    k_dft2<<<gDft, 256, 0, stream>>>(Yr, Yi, twR, twI, d200R, d200I, FqR, FqI);
    k_dft1_real<<<gDft, 256, 0, stream>>>(kk, ch0, d250R, d250I, Yr, Yi);
    k_dft2<<<gDft, 256, 0, stream>>>(Yr, Yi, twR, twI, d200R, d200I, FkR, FkI);
    k_xspec<<<gb(NN), TB, 0, stream>>>(FqR, FqI, FkR, FkI, Sr, Si);
  }

  // ---- inverse DFT -> corr, top-k, softmax, roll-aggregate ----
  k_idft1<<<L1F, 256, 0, stream>>>(Sr, Si, d250R, d250I, Y2r, Y2i);
  k_idft2<<<L1F, 256, 0, stream>>>(Y2r, Y2i, twR, twI, d200R, d200I, corr);
  k_topk_softmax<<<1, 1024, 0, stream>>>(corr, tIdx, tW);
  k_rollagg<<<NN, 128, 0, stream>>>(v, tIdx, tW, bufA);   // bufA = autocorrelation output

  // ---- output projection ----
  k_cvt_bf16<<<gb(NN * HIDF), TB, 0, stream>>>(bufA, hb, NN * HIDF);
  k_gemm_bf16<<<rowBlocks, 32, 0, stream>>>(hb, pwo, bufB, bo, HIDF);  // bufB = h2

  // ---- GraphConv2 -> d_out ----
  k_scale_to_bf16<<<gb(NN * HIDF), TB, 0, stream>>>(bufB, degOut, xsb, NN * HIDF, HIDF);
  k_gemm_bf16<<<rowBlocks, 32, 0, stream>>>(xsb, pw2, bufA, nullptr, HIDF);
  k_fill_f32<<<gb(NN * HIDF), TB, 0, stream>>>(kk, NN * HIDF, 0.f);    // reuse k as agg2
  k_scatter_edges<<<EE, 128, 0, stream>>>(bufA, src, dst, kk);
  k_postagg_f32<<<gb(NN * HIDF), TB, 0, stream>>>(kk, degIn, b2, out, NN * HIDF);

  (void)n_in; (void)in_sizes; (void)out_size; (void)ws_size;
}
